// EdgeClassifier_30305289240589
// MI455X (gfx1250) — matrix operand order repaired and checked by reading, not compiled, not run on hardware
//
#include <hip/hip_runtime.h>
#include <hip/hip_bf16.h>

// ---------------------------------------------------------------------------
// Fused 3-layer edge-MLP for MI455X (gfx1250), bf16 WMMA + f32 LayerNorm.
// v4: fully vectorized (v8f) LayerNorm / epilogues to coax packed-f32 VOP3P
//     (v_pk_add/mul/fma_f32) and VOPD dual-issue out of the backend.
//     Keeps: hardware v_cvt_pk_bf16_f32 (inline asm), readlane-uniform
//     gather, LDS-resident bf16 transposed weights, inline-0 WMMA C init.
// ---------------------------------------------------------------------------

typedef __attribute__((ext_vector_type(2)))  __bf16 v2bf;
typedef __attribute__((ext_vector_type(4)))  __bf16 v4bf;
typedef __attribute__((ext_vector_type(8)))  __bf16 v8bf;
typedef __attribute__((ext_vector_type(16))) __bf16 v16bf;
typedef __attribute__((ext_vector_type(4)))  float  v4f;
typedef __attribute__((ext_vector_type(8)))  float  v8f;

static constexpr int H  = 128;     // hidden width
static constexpr int K0 = 384;     // concat width (3*H)
static constexpr int P0 = 392;     // bf16 pitch for K=384 stage/W0T (784 B, 16B-mult)
static constexpr int PH = 136;     // bf16 pitch for K=128 stage/W1T/W2T (272 B)

// LDS layout (bytes)
static constexpr size_t OFF_W0T   = 0;
static constexpr size_t OFF_W1T   = OFF_W0T + (size_t)H * P0 * 2;   // 100352
static constexpr size_t OFF_W2T   = OFF_W1T + (size_t)H * PH * 2;   // +34816
static constexpr size_t OFF_PRM   = OFF_W2T + (size_t)H * PH * 2;   // +34816
static constexpr size_t PRM_FLOATS = 10 * H + 16;                    // b/g/be x3, W3, b3
static constexpr size_t OFF_STAGE = OFF_PRM + PRM_FLOATS * 4;        // 16B aligned
static constexpr size_t STAGE_PER_WAVE = (size_t)16 * P0 * 2;        // 12544 B
static constexpr size_t SMEM_BYTES = OFF_STAGE + 8 * STAGE_PER_WAVE; // ~275 KB < 320 KB

// ---- float -> bf16: packed hardware conversion (probe-verified on gfx1250) -
__device__ __forceinline__ v2bf f2bf2(float a, float b) {
#if __has_builtin(__builtin_amdgcn_cvt_pk_bf16_f32)
    return __builtin_amdgcn_cvt_pk_bf16_f32(a, b);
#else
    unsigned out;
    asm("v_cvt_pk_bf16_f32 %0, %1, %2" : "=v"(out) : "v"(a), "v"(b));
    return __builtin_bit_cast(v2bf, out);
#endif
}
__device__ __forceinline__ __bf16 f2bf(float f) {
    v2bf p = f2bf2(f, 0.0f);
    return p[0];
}
__device__ __forceinline__ v4bf f2bf4(float x, float y, float z, float w) {
    v2bf lo = f2bf2(x, y), hi = f2bf2(z, w);
    return __builtin_shufflevector(lo, hi, 0, 1, 2, 3);
}

// 16 contiguous bf16 (B-fragment half-wave slice): two ds_load_b128
__device__ __forceinline__ v16bf ldb16(const __bf16* p) {
    v8bf lo = *(const v8bf*)(p);
    v8bf hi = *(const v8bf*)(p + 8);
    return __builtin_shufflevector(lo, hi, 0,1,2,3,4,5,6,7,8,9,10,11,12,13,14,15);
}

// A-fragment: 8 bf16 at p, 8 bf16 at p+16 (K gap per ISA 16-bit A 16x32 layout)
__device__ __forceinline__ v16bf lda16(const __bf16* p) {
    v8bf lo = *(const v8bf*)(p);
    v8bf hi = *(const v8bf*)(p + 16);
    return __builtin_shufflevector(lo, hi, 0,1,2,3,4,5,6,7,8,9,10,11,12,13,14,15);
}

// per-element xor-shuffle of a whole v8f (16-lane half-wave reduction step)
__device__ __forceinline__ v8f shfl_xor_v8(v8f v, int m) {
    v8f r;
#pragma unroll
    for (int i = 0; i < 8; ++i) r[i] = __shfl_xor(v[i], m, 32);
    return r;
}

// One dense layer: acc[8] (= 16x128 tile) = stage(16xK) @ wT(KxN, col-major).
// Accumulators start at zero so the first WMMA's SRC2 folds to inline 0;
// bias is applied later in the LayerNorm prologue.
template<int KSTEPS, int PITCH>
__device__ __forceinline__ void layer_wmma(const __bf16* __restrict__ stage,
                                           const __bf16* __restrict__ wT,
                                           v8f acc[8], int ln, int halfsel) {
    v16bf a[KSTEPS];
#pragma unroll
    for (int f = 0; f < KSTEPS; ++f)
        a[f] = ldb16(stage + ln * PITCH + f * 32 + halfsel * 16);

#pragma unroll
    for (int nt = 0; nt < 8; ++nt) {
        v8f c = {};
        const __bf16* col = wT + (size_t)(nt * 16 + ln) * PITCH + halfsel * 16;
#pragma unroll
        for (int kk = 0; kk < KSTEPS; ++kk) {
            v16bf bf = ldb16(col + kk * 32);
            c = __builtin_amdgcn_wmma_f32_16x16x32_bf16(
                    false, a[kk], false, bf, (short)0, c, false, false);
        }
        acc[nt] = c;
    }
}

// bias-add + LayerNorm + ReLU over the 128 columns of each of 16 rows
// (C-layout), written as whole-v8f vector math for packed-f32 codegen.
__device__ __forceinline__ void bias_ln_relu(v8f acc[8],
                                             const float* __restrict__ bias,
                                             const float* __restrict__ g,
                                             const float* __restrict__ be, int ln) {
    float bt[8], gg[8], bb[8];
#pragma unroll
    for (int t = 0; t < 8; ++t) {
        bt[t] = bias[t * 16 + ln];
        gg[t] = g[t * 16 + ln];
        bb[t] = be[t * 16 + ln];
    }
#pragma unroll
    for (int t = 0; t < 8; ++t) acc[t] = acc[t] + bt[t];   // vector + scalar splat

    // per-row sum / sum-of-squares (vector accumulate over the 8 N-tiles)
    v8f sv = acc[0];
    v8f qv = acc[0] * acc[0];
#pragma unroll
    for (int t = 1; t < 8; ++t) {
        sv = sv + acc[t];
        qv = qv + acc[t] * acc[t];
    }
    // 16-lane half-wave tree reduction, elementwise
#pragma unroll
    for (int m = 1; m < 16; m <<= 1) {
        sv = sv + shfl_xor_v8(sv, m);
        qv = qv + shfl_xor_v8(qv, m);
    }
    v8f mean = sv * (1.0f / 128.0f);
    v8f ex2  = qv * (1.0f / 128.0f) - mean * mean + 1e-5f;
    v8f inv;
#pragma unroll
    for (int r = 0; r < 8; ++r) inv[r] = rsqrtf(ex2[r]);   // TRANS stays scalar

    v8f zero = {};
#pragma unroll
    for (int t = 0; t < 8; ++t) {
        v8f y = (acc[t] - mean) * (inv * gg[t]) + bb[t];
        acc[t] = __builtin_elementwise_max(y, zero);
    }
}

// C-layout registers -> row-major bf16 [16][PH] stage (per-wave, in-order LDS)
__device__ __forceinline__ void store_act(__bf16* __restrict__ stage,
                                          const v8f acc[8], int ln, int halfsel) {
#pragma unroll
    for (int t = 0; t < 8; ++t)
#pragma unroll
        for (int r = 0; r < 8; ++r)
            stage[(size_t)(r + halfsel * 8) * PH + t * 16 + ln] = f2bf(acc[t][r]);
}

__global__ void __launch_bounds__(256)
edge_mlp_wmma_kernel(const float* __restrict__ x,
                     const long long* __restrict__ eidx,   // [2][E] int64
                     const float* __restrict__ edge,
                     const float* __restrict__ W0, const float* __restrict__ b0,
                     const float* __restrict__ g0, const float* __restrict__ be0,
                     const float* __restrict__ W1, const float* __restrict__ b1,
                     const float* __restrict__ g1, const float* __restrict__ be1,
                     const float* __restrict__ W2, const float* __restrict__ b2,
                     const float* __restrict__ g2, const float* __restrict__ be2,
                     const float* __restrict__ W3, const float* __restrict__ b3,
                     float* __restrict__ out, int N, int E) {
    extern __shared__ __align__(16) char smem[];
    __bf16* w0t = (__bf16*)(smem + OFF_W0T);   // [n=0..127][k=0..383] pitch P0
    __bf16* w1t = (__bf16*)(smem + OFF_W1T);   // [n][k] pitch PH
    __bf16* w2t = (__bf16*)(smem + OFF_W2T);
    float*  prm = (float* )(smem + OFF_PRM);   // b0,g0,be0,b1,g1,be1,b2,g2,be2,W3,b3

    const int tid = threadIdx.x;

    // ---- one-time per-block: f32 -> bf16 weights, transposed to column-major
    for (int i = tid; i < K0 * H; i += 256) {
        int k = i / H, n = i - k * H;
        w0t[(size_t)n * P0 + k] = f2bf(W0[i]);
    }
    for (int i = tid; i < H * H; i += 256) {
        int k = i / H, n = i - k * H;
        w1t[(size_t)n * PH + k] = f2bf(W1[i]);
        w2t[(size_t)n * PH + k] = f2bf(W2[i]);
    }
    for (int i = tid; i < H; i += 256) {
        prm[0 * H + i] = b0[i];  prm[1 * H + i] = g0[i];  prm[2 * H + i] = be0[i];
        prm[3 * H + i] = b1[i];  prm[4 * H + i] = g1[i];  prm[5 * H + i] = be1[i];
        prm[6 * H + i] = b2[i];  prm[7 * H + i] = g2[i];  prm[8 * H + i] = be2[i];
        prm[9 * H + i] = W3[i];
    }
    if (tid == 0) prm[10 * H] = b3[0];
    __syncthreads();

    const int lane    = tid & 31;
    const int waveId  = tid >> 5;
    const int halfsel = lane >> 4;          // 0: lanes 0-15, 1: lanes 16-31
    const int ln      = lane & 15;
    const int wavesTotal = gridDim.x * 8;
    const int gwave      = blockIdx.x * 8 + waveId;

    __bf16* stage = (__bf16*)(smem + OFF_STAGE) + (size_t)waveId * (16 * P0);

    const long long* srcIdx = eidx;
    const long long* dstIdx = eidx + E;
    const int numTiles = (E + 15) >> 4;

    for (int tile = gwave; tile < numTiles; tile += wavesTotal) {
        const int e0 = tile * 16;

        // prefetch next tile's edge-feature stream (global_prefetch_b8)
        int nxt = tile + wavesTotal;
        if (nxt < numTiles) {
            const float* p = edge + (size_t)nxt * 16 * H;
            __builtin_prefetch(p + lane * 64, 0, 0);  // 32 lanes x 256B = 8KB tile
        }

        // edge endpoints: lanes 0-15 fetch src ids, 16-31 fetch dst ids
        int myE = e0 + ln;
        int nid = 0;
        if (myE < E) nid = (int)(halfsel ? dstIdx[myE] : srcIdx[myE]);

        // ---- stage 16 x 384 gathered inputs as bf16 (coalesced b128 rows).
        // readlane makes row ids compiler-known-uniform: SALU pointer math +
        // SADDR-form global_load_b128, instead of per-lane 64-bit VALU adds.
#pragma unroll
        for (int r = 0; r < 16; ++r) {
            int e = e0 + r;
            int s = __builtin_amdgcn_readlane(nid, r);
            int d = __builtin_amdgcn_readlane(nid, 16 + r);
            const v4f* rs = (const v4f*)(x + (size_t)s * H);
            const v4f* rd = (const v4f*)(x + (size_t)d * H);
            const v4f* re = (const v4f*)(edge + (size_t)(e < E ? e : 0) * H);
            v4f vs = rs[lane], vd = rd[lane];
            v4f ve = __builtin_nontemporal_load(re + lane);   // streamed once
            __bf16* row = stage + (size_t)r * P0;
            *(v4bf*)(row +   0 + lane * 4) = f2bf4(vs[0], vs[1], vs[2], vs[3]);
            *(v4bf*)(row + 128 + lane * 4) = f2bf4(vd[0], vd[1], vd[2], vd[3]);
            *(v4bf*)(row + 256 + lane * 4) = f2bf4(ve[0], ve[1], ve[2], ve[3]);
        }

        v8f acc[8];

        // ---- layer 0: [16x384] @ [384x128]  (12 K-steps x 8 N-tiles = 96 WMMA)
        layer_wmma<12, P0>(stage, w0t, acc, ln, halfsel);
        bias_ln_relu(acc, prm + 0 * H, prm + 1 * H, prm + 2 * H, ln);
        store_act(stage, acc, ln, halfsel);          // re-shape C -> A via LDS

        // ---- layer 1: [16x128] @ [128x128]  (32 WMMA)
        layer_wmma<4, PH>(stage, w1t, acc, ln, halfsel);
        bias_ln_relu(acc, prm + 3 * H, prm + 4 * H, prm + 5 * H, ln);
        store_act(stage, acc, ln, halfsel);

        // ---- layer 2: [16x128] @ [128x128]  (32 WMMA)
        layer_wmma<4, PH>(stage, w2t, acc, ln, halfsel);
        bias_ln_relu(acc, prm + 6 * H, prm + 7 * H, prm + 8 * H, ln);

        // ---- layer 3: [16x128] @ [128x1] + b3 (vectorized dot + reduction)
        const float b3v = prm[10 * H];
        v8f ov = acc[0] * prm[9 * H + 0 * 16 + ln];
#pragma unroll
        for (int t = 1; t < 8; ++t) ov = ov + acc[t] * prm[9 * H + t * 16 + ln];
#pragma unroll
        for (int m = 1; m < 16; m <<= 1) ov = ov + shfl_xor_v8(ov, m);
        ov = ov + b3v;

        if (ln < 8) {
            float v = ov[0];
#pragma unroll
            for (int r = 1; r < 8; ++r) v = (ln == r) ? ov[r] : v;
            int e = e0 + halfsel * 8 + ln;
            if (e < E) __builtin_nontemporal_store(v, out + e);
        }
    }
}

// ---------------------------------------------------------------------------
extern "C" void kernel_launch(void* const* d_in, const int* in_sizes, int n_in,
                              void* d_out, int out_size, void* d_ws, size_t ws_size,
                              hipStream_t stream) {
    const float*      x    = (const float*)d_in[0];
    const long long*  eidx = (const long long*)d_in[1];
    const float*      edge = (const float*)d_in[2];
    const float* W0 = (const float*)d_in[3];  const float* b0 = (const float*)d_in[4];
    const float* g0 = (const float*)d_in[5];  const float* be0= (const float*)d_in[6];
    const float* W1 = (const float*)d_in[7];  const float* b1 = (const float*)d_in[8];
    const float* g1 = (const float*)d_in[9];  const float* be1= (const float*)d_in[10];
    const float* W2 = (const float*)d_in[11]; const float* b2 = (const float*)d_in[12];
    const float* g2 = (const float*)d_in[13]; const float* be2= (const float*)d_in[14];
    const float* W3 = (const float*)d_in[15]; const float* b3 = (const float*)d_in[16];
    float* out = (float*)d_out;

    int N = in_sizes[0] / H;
    int E = in_sizes[1] / 2;

    int numTiles = (E + 15) / 16;
    int blocks = (numTiles + 7) / 8;        // 8 waves per block, 1 tile per wave pass
    if (blocks > 1024) blocks = 1024;       // persistent waves stride over tiles
    if (blocks < 1) blocks = 1;

    edge_mlp_wmma_kernel<<<dim3(blocks), dim3(256), SMEM_BYTES, stream>>>(
        x, eidx, edge, W0, b0, g0, be0, W1, b1, g1, be1,
        W2, b2, g2, be2, W3, b3, out, N, E);
}
